// DVAE_Generative_68212670595204
// MI455X (gfx1250) — matrix operand order repaired and checked by reading, not compiled, not run on hardware
//
#include <hip/hip_runtime.h>
#include <hip/hip_bf16.h>

// ---------------------------------------------------------------------------
// DVAE generative pass for MI455X (gfx1250), wave32 + WMMA bf16.
//   B=512 T=1024 XD=64 UD=32 ZD=32 HD=512 MLP_H=128
// All GEMMs use v_wmma_f32_16x16x32_bf16 (bf16 in, f32 accum).
// GRU state is double-buffered bf16 in LDS; per-lane h_prev carried in regs.
// ---------------------------------------------------------------------------

typedef __bf16 bf16;
typedef bf16  v16bf __attribute__((ext_vector_type(16)));
typedef float v8f   __attribute__((ext_vector_type(8)));

namespace {
constexpr int  NB  = 512;    // batch
constexpr int  NT  = 1024;   // time
constexpr int  XDi = 64;
constexpr int  UDi = 32;
constexpr int  ZDi = 32;
constexpr int  HDi = 512;
constexpr int  G3  = 3 * HDi;   // 1536
constexpr int  MH  = 128;       // MLP hidden
constexpr long BT  = (long)NB * NT;
constexpr int  HPAD = 520;      // bf16 row pitch for LDS h (bank spread)
}

// --------------------------- WMMA fragment helpers -------------------------
// A fragment (M=16 x K=32, bf16), source row-major, base = &src[row0*ld + k0].
// Layout (ISA 7.12.2): lane m=l&15, half=l>>4; f[0..7]=A[m][k0+half*8+0..7],
//                      f[8..15]=A[m][k0+16+half*8+0..7].
__device__ __forceinline__ v16bf load_a_bf16(const bf16* __restrict__ base, int ld) {
  const int l  = threadIdx.x & 31;
  const bf16* p = base + (long)(l & 15) * ld + ((l >> 4) << 3);
  union { uint4 q; bf16 h[8]; } lo, hi;
  lo.q = *(const uint4*)(p);
  hi.q = *(const uint4*)(p + 16);
  v16bf f;
#pragma unroll
  for (int i = 0; i < 8; ++i) { f[i] = lo.h[i]; f[8 + i] = hi.h[i]; }
  return f;
}

// B fragment (K=32 x N=16, bf16) with B[k][n] = W[n][k], W row-major [N][K].
// Layout: lane n=l&15, half=l>>4; f[0..7] = W[n][k0 + half*8 + 0..7], f[8..15] = W[n][k0 + 16 + half*8 + 0..7].
__device__ __forceinline__ v16bf load_b_bf16(const bf16* __restrict__ base, int ld) {
  const int l  = threadIdx.x & 31;
  const bf16* p = base + (long)(l & 15) * ld + ((l >> 4) << 3);
  union { uint4 q[2]; bf16 h[16]; } u;
  u.q[0] = *(const uint4*)(p);
  u.q[1] = *(const uint4*)(p + 16);
  v16bf f;
#pragma unroll
  for (int i = 0; i < 16; ++i) f[i] = u.h[i];
  return f;
}

__device__ __forceinline__ v8f wmma_bf16(v16bf a, v16bf b, v8f c) {
  return __builtin_amdgcn_wmma_f32_16x16x32_bf16(false, a, false, b, (short)0, c,
                                                 false, false);
}

__device__ __forceinline__ v16bf zero_frag() {
  v16bf f;
#pragma unroll
  for (int i = 0; i < 16; ++i) f[i] = (bf16)0.0f;
  return f;
}

__device__ __forceinline__ float sigm(float x) { return 1.0f / (1.0f + __expf(-x)); }

// --------------------------- kernel 0: fp32 -> bf16 ------------------------
__global__ void k_cvt(const float* __restrict__ s, bf16* __restrict__ d, long n) {
  long i = (long)blockIdx.x * blockDim.x + threadIdx.x;
  if (i < n) d[i] = (bf16)s[i];
}

// ------------------- kernel 1: xp = x_shift @ W_ih^T + b_ih ----------------
// M = BT (tiles of 16 rows), N = 1536, K = 64. 8 waves, 12 N-tiles each.
// Teacher forcing: row (b,t) reads x[b][t-1]; t==0 rows are zero.
__global__ __launch_bounds__(256) void k_xproj(const float* __restrict__ x,
                                               const bf16*  __restrict__ Wih,
                                               const float* __restrict__ bih,
                                               bf16* __restrict__ xp) {
  const long r0   = (long)blockIdx.x * 16;
  const int  wave = threadIdx.x >> 5, lane = threadIdx.x & 31;
  const int  m_l  = lane & 15, hf = lane >> 4;

  const long r = r0 + m_l;
  const long b = r >> 10;
  const int  t = (int)(r & 1023);
  v16bf a0 = zero_frag(), a1 = zero_frag();
  if (t != 0) {
    const float* p = x + ((b << 10) + (t - 1)) * (long)XDi;
#pragma unroll
    for (int i = 0; i < 8; ++i) {
      a0[i]     = (bf16)p[hf * 8 + i];
      a0[8 + i] = (bf16)p[16 + hf * 8 + i];
      a1[i]     = (bf16)p[32 + hf * 8 + i];
      a1[8 + i] = (bf16)p[48 + hf * 8 + i];
    }
  }

  for (int tt = 0; tt < 12; ++tt) {
    const int nt = wave * 12 + tt;                 // N tile 0..95
    const bf16* wrow = Wih + (long)nt * 16 * XDi;  // W_ih row-major [1536][64]
    v8f acc = {};
    acc = wmma_bf16(a0, load_b_bf16(wrow,      XDi), acc);
    acc = wmma_bf16(a1, load_b_bf16(wrow + 32, XDi), acc);
    const int   col  = nt * 16 + m_l;
    const float bias = bih[col];
    bf16* orow = xp + r0 * (long)G3 + col;
#pragma unroll
    for (int j = 0; j < 8; ++j) {
      const int m = j + (hf << 3);
      orow[(long)m * G3] = (bf16)(acc[j] + bias);
    }
  }
}

// ------------------------- kernel 2: GRU time scan -------------------------
// 32 blocks x 512 threads (16 waves). Block owns batch rows [b0, b0+16) for
// all T steps. h is double-buffered *bf16* in LDS (A-operand-ready, no cvt on
// the critical path); each lane's fp32 h_prev for its 16 owned (m,col) cells
// is carried in registers (it equals its own h_new from the previous step).
// Wave w owns hidden columns [32w, 32w+32) in all three gates.
// Per step: hp = h @ W_hh^T + b_hh via 16x6 WMMAs/wave, W_hh streamed from L2
// (1.5 MB bf16, permanently L2-resident; no intra-WG reuse => no LDS staging).
__global__ __launch_bounds__(512) void k_gru(const bf16*  __restrict__ xp,
                                             const bf16*  __restrict__ Whh,
                                             const float* __restrict__ bhh,
                                             bf16* __restrict__ out_h) {
  __shared__ __align__(16) bf16 hsm[2][16][HPAD];
  const int  tid  = threadIdx.x, wave = tid >> 5, lane = tid & 31;
  const int  m_l  = lane & 15, hf = lane >> 4;
  const long b0   = (long)blockIdx.x * 16;

  for (int i = tid; i < 16 * HPAD; i += 512) (&hsm[0][0][0])[i] = (bf16)0.0f;

  // per-lane gate biases for the 6 owned column groups (g*2+p2)
  float bh[6];
#pragma unroll
  for (int g = 0; g < 3; ++g) {
    bh[g * 2 + 0] = bhh[g * HDi + wave * 32 + m_l];
    bh[g * 2 + 1] = bhh[g * HDi + wave * 32 + 16 + m_l];
  }

  // running per-row pointers for xp reads and out_h writes (advance each step)
  const bf16* xpp[8];
  bf16*       ohp[8];
#pragma unroll
  for (int j = 0; j < 8; ++j) {
    const long m = b0 + j + (hf << 3);
    xpp[j] = xp    + (m << 10) * (long)G3;
    ohp[j] = out_h + (m << 10) * (long)HDi;
  }

  float hreg[2][8];   // fp32 h_prev for owned cells
#pragma unroll
  for (int p2 = 0; p2 < 2; ++p2)
#pragma unroll
    for (int j = 0; j < 8; ++j) hreg[p2][j] = 0.0f;

  __syncthreads();

  int cur = 0;
  for (int t = 0; t < NT; ++t) {
    // ---- issue the 48 gate-input loads early; latency hides under WMMAs ----
    float xr[2][8], xz[2][8], xn[2][8];
#pragma unroll
    for (int p2 = 0; p2 < 2; ++p2) {
      const int col = wave * 32 + p2 * 16 + m_l;
#pragma unroll
      for (int j = 0; j < 8; ++j) {
        xr[p2][j] = (float)xpp[j][col];
        xz[p2][j] = (float)xpp[j][col + HDi];
        xn[p2][j] = (float)xpp[j][col + 2 * HDi];
      }
    }

    v8f acc[6];
#pragma unroll
    for (int i = 0; i < 6; ++i) { v8f z = {}; acc[i] = z; }

    for (int k0 = 0; k0 < HDi; k0 += 32) {
      const v16bf a = load_a_bf16(&hsm[cur][0][k0], HPAD);   // 2x ds_load_b128
#pragma unroll
      for (int g = 0; g < 3; ++g) {
#pragma unroll
        for (int p2 = 0; p2 < 2; ++p2) {
          const bf16* wrow = Whh + ((long)(g * HDi + wave * 32 + p2 * 16)) * HDi + k0;
          acc[g * 2 + p2] = wmma_bf16(a, load_b_bf16(wrow, HDi), acc[g * 2 + p2]);
        }
      }
    }

    // ---- gates + h update (wave-local; writes go to the other LDS buffer) --
#pragma unroll
    for (int p2 = 0; p2 < 2; ++p2) {
      const int col = wave * 32 + p2 * 16 + m_l;
#pragma unroll
      for (int j = 0; j < 8; ++j) {
        const int   m  = j + (hf << 3);
        const float rg = sigm(xr[p2][j] + acc[0 + p2][j] + bh[0 + p2]);
        const float zg = sigm(xz[p2][j] + acc[2 + p2][j] + bh[2 + p2]);
        const float nn = tanhf(xn[p2][j] + rg * (acc[4 + p2][j] + bh[4 + p2]));
        const float hv = (1.0f - zg) * nn + zg * hreg[p2][j];
        hreg[p2][j] = hv;
        const bf16 hb = (bf16)hv;
        hsm[cur ^ 1][m][col] = hb;
        ohp[j][(long)t * HDi + col] = hb;
      }
    }
#pragma unroll
    for (int j = 0; j < 8; ++j) xpp[j] += G3;   // advance to next time step
    __builtin_prefetch(xpp[0] + wave * 32 + m_l, 0, 1);   // warm next xp slice
    __syncthreads();    // single barrier: buf cur^1 fully written, cur free
    cur ^= 1;
  }
}

// ---------------- kernel 3: z-MLP + reparameterized sample -----------------
// Per 16-row tile: a1 = relu([h,u] @ W1^T + b1)  (K=544, N=128, 8 waves)
//                  zp = a1 @ W2^T + b2           (K=128, N=64, waves 0..3)
//                  z  = mu + exp(log_sigma)*eps
__global__ __launch_bounds__(256) void k_dz(const bf16*  __restrict__ hbf,
                                            const bf16*  __restrict__ ubf,
                                            const float* __restrict__ eps,
                                            const bf16*  __restrict__ W1,
                                            const float* __restrict__ b1,
                                            const bf16*  __restrict__ W2,
                                            const float* __restrict__ b2,
                                            float* __restrict__ z_out,
                                            bf16*  __restrict__ z_bf) {
  __shared__ __align__(16) bf16  a1s[16][136];
  __shared__ __align__(16) float zps[16][68];
  const int  tid = threadIdx.x, wave = tid >> 5, lane = tid & 31;
  const int  m_l = lane & 15, hf = lane >> 4;
  const long r0  = (long)blockIdx.x * 16;
  const int  ld1 = HDi + UDi;   // 544

  v8f acc = {};
  const bf16* w1row = W1 + (long)wave * 16 * ld1;
  for (int k0 = 0; k0 < HDi; k0 += 32)
    acc = wmma_bf16(load_a_bf16(hbf + r0 * HDi + k0, HDi),
                    load_b_bf16(w1row + k0, ld1), acc);
  acc = wmma_bf16(load_a_bf16(ubf + r0 * UDi, UDi),
                  load_b_bf16(w1row + HDi, ld1), acc);
  {
    const int   col = wave * 16 + m_l;
    const float bb  = b1[col];
#pragma unroll
    for (int j = 0; j < 8; ++j)
      a1s[j + (hf << 3)][col] = (bf16)fmaxf(acc[j] + bb, 0.0f);
  }
  __syncthreads();

  if (wave < 4) {
    v8f a2 = {};
    const bf16* w2row = W2 + (long)wave * 16 * MH;
#pragma unroll
    for (int k0 = 0; k0 < MH; k0 += 32)
      a2 = wmma_bf16(load_a_bf16(&a1s[0][k0], 136),
                     load_b_bf16(w2row + k0, MH), a2);
    const int   col = wave * 16 + m_l;
    const float bb  = b2[col];
#pragma unroll
    for (int j = 0; j < 8; ++j) zps[j + (hf << 3)][col] = a2[j] + bb;
  }
  __syncthreads();

  for (int e = tid; e < 16 * ZDi; e += 256) {
    const int  m = e >> 5, jj = e & 31;
    const long r = r0 + m;
    const float zv = zps[m][jj] + __expf(zps[m][32 + jj]) * eps[r * ZDi + jj];
    z_out[r * ZDi + jj] = zv;
    z_bf [r * ZDi + jj] = (bf16)zv;
  }
}

// ---------------------- kernel 4: x-MLP (mu_x only) ------------------------
__global__ __launch_bounds__(256) void k_dx(const bf16*  __restrict__ hbf,
                                            const bf16*  __restrict__ zbf,
                                            const bf16*  __restrict__ W1,
                                            const float* __restrict__ b1,
                                            const bf16*  __restrict__ W2,
                                            const float* __restrict__ b2,
                                            float* __restrict__ mux) {
  __shared__ __align__(16) bf16 a1s[16][136];
  const int  tid = threadIdx.x, wave = tid >> 5, lane = tid & 31;
  const int  m_l = lane & 15, hf = lane >> 4;
  const long r0  = (long)blockIdx.x * 16;
  const int  ld1 = HDi + ZDi;   // 544

  v8f acc = {};
  const bf16* w1row = W1 + (long)wave * 16 * ld1;
  for (int k0 = 0; k0 < HDi; k0 += 32)
    acc = wmma_bf16(load_a_bf16(hbf + r0 * HDi + k0, HDi),
                    load_b_bf16(w1row + k0, ld1), acc);
  acc = wmma_bf16(load_a_bf16(zbf + r0 * ZDi, ZDi),
                  load_b_bf16(w1row + HDi, ld1), acc);
  {
    const int   col = wave * 16 + m_l;
    const float bb  = b1[col];
#pragma unroll
    for (int j = 0; j < 8; ++j)
      a1s[j + (hf << 3)][col] = (bf16)fmaxf(acc[j] + bb, 0.0f);
  }
  __syncthreads();

  if (wave < 4) {                    // only rows 0..63 of W2 => mu_x
    v8f a2 = {};
    const bf16* w2row = W2 + (long)wave * 16 * MH;
#pragma unroll
    for (int k0 = 0; k0 < MH; k0 += 32)
      a2 = wmma_bf16(load_a_bf16(&a1s[0][k0], 136),
                     load_b_bf16(w2row + k0, MH), a2);
    const int   col = wave * 16 + m_l;       // 0..63
    const float bb  = b2[col];
#pragma unroll
    for (int j = 0; j < 8; ++j) {
      const int m = j + (hf << 3);
      mux[(r0 + m) * (long)XDi + col] = a2[j] + bb;   // mu_x ld = 64
    }
  }
}

// ------------------------------- launcher ----------------------------------
extern "C" void kernel_launch(void* const* d_in, const int* in_sizes, int n_in,
                              void* d_out, int out_size, void* d_ws, size_t ws_size,
                              hipStream_t stream) {
  (void)in_sizes; (void)n_in; (void)out_size; (void)ws_size;
  const float* x     = (const float*)d_in[0];
  const float* u     = (const float*)d_in[1];
  const float* eps   = (const float*)d_in[2];
  const float* W_ih  = (const float*)d_in[3];
  const float* W_hh  = (const float*)d_in[4];
  const float* b_ih  = (const float*)d_in[5];
  const float* b_hh  = (const float*)d_in[6];
  const float* dzW1  = (const float*)d_in[7];
  const float* dzb1  = (const float*)d_in[8];
  const float* dzW2  = (const float*)d_in[9];
  const float* dzb2  = (const float*)d_in[10];
  const float* dxW1  = (const float*)d_in[11];
  const float* dxb1  = (const float*)d_in[12];
  const float* dxW2  = (const float*)d_in[13];
  const float* dxb2  = (const float*)d_in[14];

  float* z_out   = (float*)d_out;                    // [BT,32]
  float* mux_out = (float*)d_out + BT * ZDi;         // [BT,64]

  // workspace carve-up (bf16 scratch; ~2.22 GB total)
  char* w = (char*)d_ws;
  bf16* xp     = (bf16*)w;  w += BT * (long)G3  * 2;
  bf16* hbf    = (bf16*)w;  w += BT * (long)HDi * 2;
  bf16* ubf    = (bf16*)w;  w += BT * (long)UDi * 2;
  bf16* zbf    = (bf16*)w;  w += BT * (long)UDi * 2;
  bf16* Wih_b  = (bf16*)w;  w += (long)G3 * XDi * 2;
  bf16* Whh_b  = (bf16*)w;  w += (long)G3 * HDi * 2;
  bf16* dzW1_b = (bf16*)w;  w += (long)MH * (HDi + UDi) * 2;
  bf16* dzW2_b = (bf16*)w;  w += (long)(2 * ZDi) * MH * 2;
  bf16* dxW1_b = (bf16*)w;  w += (long)MH * (HDi + ZDi) * 2;
  bf16* dxW2_b = (bf16*)w;  w += (long)(2 * XDi) * MH * 2;

  auto cvt = [&](const float* s, bf16* d, long n) {
    k_cvt<<<(int)((n + 255) / 256), 256, 0, stream>>>(s, d, n);
  };
  cvt(W_ih, Wih_b,  (long)G3 * XDi);
  cvt(W_hh, Whh_b,  (long)G3 * HDi);
  cvt(dzW1, dzW1_b, (long)MH * (HDi + UDi));
  cvt(dzW2, dzW2_b, (long)(2 * ZDi) * MH);
  cvt(dxW1, dxW1_b, (long)MH * (HDi + ZDi));
  cvt(dxW2, dxW2_b, (long)(2 * XDi) * MH);
  cvt(u,    ubf,    BT * (long)UDi);

  k_xproj<<<(int)(BT / 16), 256, 0, stream>>>(x, Wih_b, b_ih, xp);
  k_gru  <<<NB / 16,        512, 0, stream>>>(xp, Whh_b, b_hh, hbf);
  k_dz   <<<(int)(BT / 16), 256, 0, stream>>>(hbf, ubf, eps, dzW1_b, dzb1,
                                              dzW2_b, dzb2, z_out, zbf);
  k_dx   <<<(int)(BT / 16), 256, 0, stream>>>(hbf, zbf, dxW1_b, dxb1,
                                              dxW2_b, dxb2, mux_out);
}
